// AGSG_88527865905453
// MI455X (gfx1250) — matrix operand order repaired and checked by reading, compiled, not run on hardware
//
#include <hip/hip_runtime.h>
#include <math.h>

// ---------------- WMMA / TDM types (CDNA5, wave32) ----------------
typedef __attribute__((ext_vector_type(16))) __bf16 bf16x16;
typedef __attribute__((ext_vector_type(8)))  float  floatx8;
typedef unsigned int u32x4 __attribute__((ext_vector_type(4)));
typedef int          i32x4 __attribute__((ext_vector_type(4)));
typedef int          i32x8 __attribute__((ext_vector_type(8)));

#if defined(__has_builtin)
#if __has_builtin(__builtin_amdgcn_tensor_load_to_lds)
#define HAVE_TDM 1
#endif
#endif

__device__ __forceinline__ __bf16 bf_bits(unsigned short u) {
  return __builtin_bit_cast(__bf16, u);
}
__device__ __forceinline__ unsigned short f2bf_bits(float f) {
  unsigned int u = __builtin_bit_cast(unsigned int, f);
  unsigned int r = (u + 0x7FFFu + ((u >> 16) & 1u)) >> 16;  // round-nearest-even
  return (unsigned short)r;
}

// ---------------- TDM 2D tile load: global bf16 -> LDS ----------------
// D# per ISA 8.3/8.4: group0 = {count=1, lds_addr, global_addr, type=2},
// group1 = {data_size=2B, big tensor dims, tile 32x64, 48-bit dim0 stride}.
#if defined(HAVE_TDM)
__device__ __forceinline__ void tdm_load_tile(unsigned lds_off,
                                              const unsigned short* gptr,
                                              unsigned stride0_elems)
{
  unsigned long long ga = (unsigned long long)(const void*)gptr;
  const unsigned td0 = 1u << 20, td1 = 1u << 20;  // huge: tile never clipped
  u32x4 g0 = { 1u,                                   // count=1 (user mode)
               lds_off,                               // LDS byte address
               (unsigned)ga,                          // global addr lo
               (unsigned)((ga >> 32) & 0x01FFFFFFu) | 0x80000000u };  // hi, type=2
  i32x8 g1 = { (int)(1u << 16),                               // data_size code 1 (2B)
               (int)((td0 & 0xFFFFu) << 16),                  // tensor_dim0 lo
               (int)((td0 >> 16) | ((td1 & 0xFFFFu) << 16)),  // dim0 hi | dim1 lo
               (int)((td1 >> 16) | (32u << 16)),              // dim1 hi | tile_dim0=32
               64,                                            // tile_dim1=64 (dim2=0)
               (int)stride0_elems,                            // dim0 stride lo32
               0, 0 };
  i32x4 gz = { 0, 0, 0, 0 };
#if __clang_major__ >= 23
  i32x8 gz8 = { 0, 0, 0, 0, 0, 0, 0, 0 };
  __builtin_amdgcn_tensor_load_to_lds(g0, g1, gz, gz, gz8, 0);
#else
  __builtin_amdgcn_tensor_load_to_lds(g0, g1, gz, gz, 0);
#endif
}
#endif

// ---------------- block reductions (blockDim.x == 256) ----------------
__device__ __forceinline__ float blk_max(float v, float* red) {
  int t = threadIdx.x;
  red[t] = v; __syncthreads();
  for (int s = 128; s > 0; s >>= 1) {
    if (t < s) red[t] = fmaxf(red[t], red[t + s]);
    __syncthreads();
  }
  float r = red[0]; __syncthreads();
  return r;
}
__device__ __forceinline__ float blk_sum(float v, float* red) {
  int t = threadIdx.x;
  red[t] = v; __syncthreads();
  for (int s = 128; s > 0; s >>= 1) {
    if (t < s) red[t] = red[t] + red[t + s];
    __syncthreads();
  }
  float r = red[0]; __syncthreads();
  return r;
}

// -------- fragment build + 2x WMMA; A tile [m][k], B tile [n][k] (both k-contig)
__device__ __forceinline__ void mma_step(const unsigned short (*As)[32],
                                         const unsigned short (*Bs)[32],
                                         int wm, int wn, int r, int half,
                                         floatx8& acc0, floatx8& acc1)
{
  bf16x16 a0, a1, bfr;
#pragma unroll
  for (int v = 0; v < 8; ++v) {
    int ka = ((v < 4) ? (2 * v) : (2 * v + 8)) + 8 * half;   // two 16B runs / lane
    a0[2 * v]     = bf_bits(As[wm + r][ka]);
    a0[2 * v + 1] = bf_bits(As[wm + r][ka + 1]);
    a1[2 * v]     = bf_bits(As[wm + 16 + r][ka]);
    a1[2 * v + 1] = bf_bits(As[wm + 16 + r][ka + 1]);
    int kb = ((v < 4) ? (2 * v) : (2 * v + 8)) + 8 * half;   // two 16B runs / lane
    bfr[2 * v]     = bf_bits(Bs[wn + r][kb]);
    bfr[2 * v + 1] = bf_bits(Bs[wn + r][kb + 1]);
  }
  acc0 = __builtin_amdgcn_wmma_f32_16x16x32_bf16(false, a0, false, bfr,
                                                 (short)0, acc0, false, false);
  acc1 = __builtin_amdgcn_wmma_f32_16x16x32_bf16(false, a1, false, bfr,
                                                 (short)0, acc1, false, false);
}

// ---------------- bf16-WMMA GEMM, TDM double-buffered staging ----------------
// Cout = relu?( alpha * A@B (+ Cin) ); optional bf16 mirrors:
//   C16  row-major [M,N]  (A-operand layout for later GEMMs)
//   C16T transposed [N,M] (B-operand layout for later GEMMs)
// A: bf16 [M,K] row-major; Bt: bf16 [N,K] row-major (i.e. B transposed).
__global__ __launch_bounds__(256) void gemm_bf16wmma(
    const unsigned short* __restrict__ A, const unsigned short* __restrict__ Bt,
    const float* __restrict__ Cin, float* __restrict__ Cout,
    unsigned short* __restrict__ Cout16, unsigned short* __restrict__ Cout16T,
    int M, int N, int K, float alpha, int doRelu,
    long long strideA, long long strideB, long long strideC)
{
  __shared__ __align__(16) unsigned short AsBuf[2][64][32];  // [m][k]
  __shared__ __align__(16) unsigned short BsBuf[2][64][32];  // [n][k]

  const int tid  = threadIdx.x;
  const int lane = tid & 31;
  const int half = lane >> 4;
  const int r    = lane & 15;
  const int w    = tid >> 5;
  const int wm   = (w >> 2) << 5;  // 0 or 32
  const int wn   = (w & 3) << 4;   // 0,16,32,48
  const int bm   = blockIdx.y << 6;
  const int bn   = blockIdx.x << 6;

  const long long bz = blockIdx.z;
  const unsigned short* Ab  = A  + bz * strideA;
  const unsigned short* Btb = Bt + bz * strideB;
  float*          Cb   = Cout + bz * strideC;
  const float*    Ci   = Cin    ? (Cin    + bz * strideC) : nullptr;
  unsigned short* C16  = Cout16 ? (Cout16 + bz * strideC) : nullptr;
  unsigned short* C16T = Cout16T? (Cout16T+ bz * strideC) : nullptr;

  floatx8 acc0 = {};
  floatx8 acc1 = {};
  const int nIter = K >> 5;

#if defined(HAVE_TDM)
  const unsigned ldsA0 = (unsigned)(unsigned long long)(void*)&AsBuf[0][0][0];
  const unsigned ldsA1 = (unsigned)(unsigned long long)(void*)&AsBuf[1][0][0];
  const unsigned ldsB0 = (unsigned)(unsigned long long)(void*)&BsBuf[0][0][0];
  const unsigned ldsB1 = (unsigned)(unsigned long long)(void*)&BsBuf[1][0][0];
  if (w == 0) {  // prologue: DMA first tile pair into buffer 0
    tdm_load_tile(ldsA0, Ab  + (long long)bm * K, (unsigned)K);
    tdm_load_tile(ldsB0, Btb + (long long)bn * K, (unsigned)K);
  }
  for (int it = 0; it < nIter; ++it) {
    const int cur = it & 1;
    if (w == 0) {
      if (it + 1 < nIter) {  // issue next tile pair into the other buffer
        const long long k1 = (long long)(it + 1) << 5;
        tdm_load_tile(cur ? ldsA0 : ldsA1, Ab  + (long long)bm * K + k1, (unsigned)K);
        tdm_load_tile(cur ? ldsB0 : ldsB1, Btb + (long long)bn * K + k1, (unsigned)K);
        __builtin_amdgcn_s_wait_tensorcnt(2);  // only the 2 just-issued outstanding
      } else {
        __builtin_amdgcn_s_wait_tensorcnt(0);
      }
    }
    __syncthreads();                           // current buffer visible to all waves
    mma_step(AsBuf[cur], BsBuf[cur], wm, wn, r, half, acc0, acc1);
    __syncthreads();                           // reads done before buffer reuse
  }
#else
  for (int it = 0; it < nIter; ++it) {
    const int k0 = it << 5;
#pragma unroll
    for (int i = 0; i < 8; ++i) {
      int e = tid + i * 256;
      int row = e >> 5, k = e & 31;
      AsBuf[0][row][k] = Ab[(long long)(bm + row) * K + (k0 + k)];
      BsBuf[0][row][k] = Btb[(long long)(bn + row) * K + (k0 + k)];
    }
    __syncthreads();
    mma_step(AsBuf[0], BsBuf[0], wm, wn, r, half, acc0, acc1);
    __syncthreads();
  }
#endif

  // epilogue: C/D layout VGPR v -> (M = v + 8*half, N = r)
#pragma unroll
  for (int v = 0; v < 8; ++v) {
    int row0 = bm + wm + v + 8 * half;
    int row1 = row0 + 16;
    int col  = bn + wn + r;
    long long i0 = (long long)row0 * N + col;
    long long i1 = (long long)row1 * N + col;
    float v0 = acc0[v] * alpha;
    float v1 = acc1[v] * alpha;
    if (Ci) { v0 += Ci[i0]; v1 += Ci[i1]; }
    if (doRelu) { v0 = fmaxf(v0, 0.f); v1 = fmaxf(v1, 0.f); }
    Cb[i0] = v0;
    Cb[i1] = v1;
    if (C16)  { C16[i0] = f2bf_bits(v0); C16[i1] = f2bf_bits(v1); }
    if (C16T) {
      C16T[(long long)col * M + row0] = f2bf_bits(v0);
      C16T[(long long)col * M + row1] = f2bf_bits(v1);
    }
  }
}

// ---------------- in-place row softmax over 512 cols ----------------
__global__ __launch_bounds__(256) void row_softmax512(float* __restrict__ data,
                                                      int diagMod, float diagVal,
                                                      unsigned short* __restrict__ out16,
                                                      unsigned short* __restrict__ out16T)
{
  __shared__ float red[256];
  const int row = blockIdx.x;
  const int tid = threadIdx.x;
  float* p = data + (long long)row * 512;
  float v0 = p[tid];
  float v1 = p[tid + 256];
  if (diagMod > 0) {
    int dcol = row % diagMod;
    if (tid == dcol) v0 = diagVal;
    if (tid + 256 == dcol) v1 = diagVal;
  }
  float mx = blk_max(fmaxf(v0, v1), red);
  float e0 = __expf(v0 - mx), e1 = __expf(v1 - mx);
  float s = blk_sum(e0 + e1, red);
  float inv = 1.f / s;
  float r0 = e0 * inv, r1 = e1 * inv;
  p[tid] = r0;
  p[tid + 256] = r1;
  if (out16) {
    out16[(long long)row * 512 + tid] = f2bf_bits(r0);
    out16[(long long)row * 512 + tid + 256] = f2bf_bits(r1);
  }
  if (out16T) {
    out16T[(long long)tid * 512 + row] = f2bf_bits(r0);
    out16T[(long long)(tid + 256) * 512 + row] = f2bf_bits(r1);
  }
}

// ---------------- row sums of S_w ----------------
__global__ __launch_bounds__(256) void row_sum512(const float* __restrict__ mat,
                                                  float* __restrict__ out)
{
  __shared__ float red[256];
  const int row = blockIdx.x;
  const int tid = threadIdx.x;
  const float* p = mat + (long long)row * 512;
  float s = blk_sum(p[tid] + p[tid + 256], red);
  if (tid == 0) out[row] = s;
}

// ------- memory param -> bf16 transposed [N=512, C=64] (A and Bt operand) ----
__global__ void cvt_mem_kernel(const float* __restrict__ m,
                               unsigned short* __restrict__ m16T)
{
  int i = blockIdx.x * blockDim.x + threadIdx.x;  // [c=64][n=512]
  if (i >= 64 * 512) return;
  int c = i >> 9, n = i & 511;
  m16T[n * 64 + c] = f2bf_bits(m[i]);
}

// ---------------- x_sum[b,n] = sum_{c,t} x[b,c,n,t] ----------------
__global__ void xsum_kernel(const float* __restrict__ x, float* __restrict__ xsum)
{
  int i = blockIdx.x * blockDim.x + threadIdx.x;  // 0..8191
  if (i >= 16 * 512) return;
  int b = i >> 9, n = i & 511;
  const float* px = x + ((long long)b * 64 * 512 + n) * 12;
  float s = 0.f;
  for (int c = 0; c < 64; ++c) {
    const float* q = px + (long long)c * 512 * 12;
#pragma unroll
    for (int t = 0; t < 12; ++t) s += q[t];
  }
  xsum[i] = s;
}

// ------- MHSG: scale+relu+softmax(6144)+time-sum per (b,c) -> bf16 [b][n][c] --
__global__ __launch_bounds__(256) void mhsg_kernel(const float* __restrict__ x,
                                                   const float* __restrict__ rowsumS,
                                                   unsigned short* __restrict__ xws16T)
{
  __shared__ float red[256];
  const int bc  = blockIdx.x;  // b*64 + c
  const int tid = threadIdx.x;
  const float* px = x + (long long)bc * 6144;
  float v[24];
  float mx = -1e30f;
#pragma unroll
  for (int j = 0; j < 24; ++j) {
    int i = tid * 24 + j;
    // rs_flat[i] = rowsum[i % 512] + 0.8*(11 - i/512)  (faithful index mismatch)
    float mult = rowsumS[i & 511] + 0.8f * (float)(11 - (i >> 9));
    float val = fmaxf(px[i] * mult * 0.125f, 0.f);
    v[j] = val;
    mx = fmaxf(mx, val);
  }
  mx = blk_max(mx, red);
  float s = 0.f;
#pragma unroll
  for (int j = 0; j < 24; ++j) { v[j] = __expf(v[j] - mx); s += v[j]; }
  s = blk_sum(s, red);
  float inv = 1.f / s;
  float s0 = 0.f, s1 = 0.f;
#pragma unroll
  for (int j = 0; j < 12; ++j)  s0 += v[j];
#pragma unroll
  for (int j = 12; j < 24; ++j) s1 += v[j];
  const int b = bc >> 6, c = bc & 63, n0 = 2 * tid;
  // [b][n][c]: serves as BOTH A ([M,K]) and Bt ([N,K]) operand of the A_l gram
  xws16T[(long long)b * 32768 + (long long)n0 * 64 + c]       = f2bf_bits(s0 * inv);
  xws16T[(long long)b * 32768 + (long long)(n0 + 1) * 64 + c] = f2bf_bits(s1 * inv);
}

// ---------------- A_p: softmax_m relu(xsum[b,n]*(F+P)[n,m]/8) ----------------
__global__ __launch_bounds__(256) void ap_kernel(const float* __restrict__ F,
                                                 const float* __restrict__ P,
                                                 const float* __restrict__ xsum,
                                                 float* __restrict__ Ap)
{
  __shared__ float red[256];
  const int blk = blockIdx.x;  // b*512 + n
  const int n   = blk & 511;
  const int tid = threadIdx.x;
  float scale = xsum[blk] * 0.125f;
  const float* pf = F + (long long)n * 512;
  const float* pp = P + (long long)n * 512;
  float v0 = fmaxf(scale * (pf[tid] + pp[tid]), 0.f);
  float v1 = fmaxf(scale * (pf[tid + 256] + pp[tid + 256]), 0.f);
  float mx = blk_max(fmaxf(v0, v1), red);
  float e0 = __expf(v0 - mx), e1 = __expf(v1 - mx);
  float s = blk_sum(e0 + e1, red);
  float inv = 1.f / s;
  float* po = Ap + (long long)blk * 512;
  po[tid] = e0 * inv;
  po[tid + 256] = e1 * inv;
}

// ---------------- F(1) = I + S_w (f32 + transposed bf16) ----------------
__global__ void initF_kernel(const float* __restrict__ S, float* __restrict__ F,
                             unsigned short* __restrict__ F16T)
{
  int i = blockIdx.x * blockDim.x + threadIdx.x;
  if (i >= 512 * 512) return;
  int row = i >> 9, col = i & 511;
  float v = S[i] + ((row == col) ? 1.f : 0.f);
  F[i] = v;
  F16T[(long long)col * 512 + row] = f2bf_bits(v);  // Bt layout (F used only as B)
}

// ---------------- driver ----------------
extern "C" void kernel_launch(void* const* d_in, const int* in_sizes, int n_in,
                              void* d_out, int out_size, void* d_ws, size_t ws_size,
                              hipStream_t stream)
{
  const float* x      = (const float*)d_in[0];  // [16,64,512,12]
  const float* memory = (const float*)d_in[1];  // [64,512]
  float* out = (float*)d_out;                   // A_p [16,512,512] ++ A_l [16,512,512]
  float* ws  = (float*)d_ws;

  const long long NN = 512LL * 512LL;
  float* S       = ws;
  float* F       = S + NN;
  float* P       = F + NN;
  float* rowsumS = P + NN;             // 512
  float* xsum    = rowsumS + 512;      // 16*512
  unsigned short* u16 = (unsigned short*)(xsum + 16 * 512);
  unsigned short* S16    = u16; u16 += NN;   // S row-major   (A operand)
  unsigned short* S16T   = u16; u16 += NN;   // S transposed  (Bt operand)
  unsigned short* F16T   = u16; u16 += NN;   // F ping/pong   (Bt operand)
  unsigned short* Ft16T  = u16; u16 += NN;
  unsigned short* P16    = u16; u16 += NN;   // P ping/pong   (A operand)
  unsigned short* Pt16   = u16; u16 += NN;
  unsigned short* P16T   = u16; u16 += NN;   // P ping/pong   (Bt operand)
  unsigned short* Pt16T  = u16; u16 += NN;
  unsigned short* mem16T = u16; u16 += 64 * 512;        // [512,64]
  unsigned short* xws16T = u16; u16 += 16LL * 64 * 512; // [b][512][64]

  float* Ap = out;
  float* Al = out + 16 * NN;
  float* Ft = Ap;            // f32 ping-pong scratch inside (still dead) A_p region
  float* Pt = Ap + NN;

  const float inv_sqrt_c = 0.125f;
  dim3 blk(256);
  dim3 g1(8, 8, 1);

  // 0) memory -> bf16 [N,C] (serves as A and Bt of the gram)
  cvt_mem_kernel<<<128, blk, 0, stream>>>(memory, mem16T);
  // 1) G = relu(memory^T @ memory)   [512x512, K=64]
  gemm_bf16wmma<<<g1, blk, 0, stream>>>(mem16T, mem16T, nullptr, S, nullptr, nullptr,
                                        512, 512, 64, 1.0f, 1, 0, 0, 0);
  // 2) diag <- 0.1, row softmax -> S_w (+ bf16 mirrors, both layouts)
  row_softmax512<<<512, blk, 0, stream>>>(S, 512, 0.1f, S16, S16T);
  // 3) row sums of S_w
  row_sum512<<<512, blk, 0, stream>>>(S, rowsumS);
  // 4) F(1)=I+S_w ; 5) P(1)=S_w@S_w
  initF_kernel<<<1024, blk, 0, stream>>>(S, F, F16T);
  gemm_bf16wmma<<<g1, blk, 0, stream>>>(S16, S16T, nullptr, P, P16, P16T,
                                        512, 512, 512, 1.0f, 0, 0, 0, 0);
  // 6) 8 doubling steps: F' = F + P@F ; P' = P@P   (n: 1->3->...->511)
  float* fs = F;  float* fd = Ft;  float* ps = P;  float* pd = Pt;
  unsigned short* fs16T = F16T; unsigned short* fd16T = Ft16T;
  unsigned short* ps16  = P16;  unsigned short* pd16  = Pt16;
  unsigned short* ps16T = P16T; unsigned short* pd16T = Pt16T;
  for (int s = 0; s < 8; ++s) {
    gemm_bf16wmma<<<g1, blk, 0, stream>>>(ps16, fs16T, fs, fd, nullptr, fd16T,
                                          512, 512, 512, 1.0f, 0, 0, 0, 0);
    gemm_bf16wmma<<<g1, blk, 0, stream>>>(ps16, ps16T, nullptr, pd, pd16, pd16T,
                                          512, 512, 512, 1.0f, 0, 0, 0, 0);
    float* t;
    t = fs; fs = fd; fd = t;
    t = ps; ps = pd; pd = t;
    unsigned short* t16;
    t16 = fs16T; fs16T = fd16T; fd16T = t16;
    t16 = ps16;  ps16  = pd16;  pd16  = t16;
    t16 = ps16T; ps16T = pd16T; pd16T = t16;
  }
  // after 8 swaps fs==F, ps==P ; sup_sum = F(511)+P(511) folded into ap_kernel

  // 7) x_sum ; 8) MHSG softmax + time-sum -> bf16 [b][n][c]
  xsum_kernel<<<32, blk, 0, stream>>>(x, xsum);
  mhsg_kernel<<<1024, blk, 0, stream>>>(x, rowsumS, xws16T);

  // 9) A_l pre-activation: relu( X_b^T X_b / 8 ), batched over b (K=64)
  gemm_bf16wmma<<<dim3(8, 8, 16), blk, 0, stream>>>(
      xws16T, xws16T, nullptr, Al, nullptr, nullptr, 512, 512, 64, inv_sqrt_c, 1,
      32768LL, 32768LL, NN);
  // 10) row softmax A_l (in place)
  row_softmax512<<<16 * 512, blk, 0, stream>>>(Al, 0, 0.f, nullptr, nullptr);
  // 11) A_p (overwrites Ft/Pt scratch with the real output)
  ap_kernel<<<16 * 512, blk, 0, stream>>>(fs, ps, xsum, Ap);
}
